// InfoNCE_12051678232817
// MI455X (gfx1250) — matrix operand order repaired and checked by reading, not compiled, not run on hardware
//
#include <hip/hip_runtime.h>
#include <hip/hip_bf16.h>
#include <math.h>

typedef __attribute__((ext_vector_type(16))) _Float16 v16h;
typedef __attribute__((ext_vector_type(8)))  _Float16 v8h;
typedef __attribute__((ext_vector_type(8)))  float    v8f;

#define B 512
#define D 512
#define EPS 1e-6f

// ---------------------------------------------------------------- reductions
template<int N>
__device__ __forceinline__ float block_sum(float v, float* red) {
    int t = threadIdx.x;
    red[t] = v; __syncthreads();
#pragma unroll
    for (int s = N / 2; s > 0; s >>= 1) {
        if (t < s) red[t] += red[t + s];
        __syncthreads();
    }
    float r = red[0]; __syncthreads();
    return r;
}

template<int N>
__device__ __forceinline__ float block_min(float v, float* red) {
    int t = threadIdx.x;
    red[t] = v; __syncthreads();
#pragma unroll
    for (int s = N / 2; s > 0; s >>= 1) {
        if (t < s) red[t] = fminf(red[t], red[t + s]);
        __syncthreads();
    }
    float r = red[0]; __syncthreads();
    return r;
}

template<int N>
__device__ __forceinline__ float block_max(float v, float* red) {
    int t = threadIdx.x;
    red[t] = v; __syncthreads();
#pragma unroll
    for (int s = N / 2; s > 0; s >>= 1) {
        if (t < s) red[t] = fmaxf(red[t], red[t + s]);
        __syncthreads();
    }
    float r = red[0]; __syncthreads();
    return r;
}

// ------------------------------------------------- kernel 0: per-row stats
// blocks 0..511  -> rows of (mu1, sigma1); blocks 512..1023 -> (mu2, sigma2)
// Produces: l2-normalized rows (f32 + f16), sum(log(sigma+eps)), mean(sigma),
// and ||n||^2 (== sumsq*inv^2, matching the reference's Gram expansion).
__global__ void rowstats_kernel(const float* __restrict__ mu1, const float* __restrict__ sg1,
                                const float* __restrict__ mu2, const float* __restrict__ sg2,
                                float* n1f, float* n2f, _Float16* n1h, _Float16* n2h,
                                float* logdet1, float* logdet2,
                                float* smean1, float* smean2,
                                float* sq1, float* sq2) {
    __shared__ float red[256];
    const int row  = blockIdx.x & (B - 1);
    const bool s2  = blockIdx.x >= B;
    const float* mu = (s2 ? mu2 : mu1) + row * D;
    const float* sg = (s2 ? sg2 : sg1) + row * D;

    float v[2];
    float lsq = 0.f, llog = 0.f, lsig = 0.f;
#pragma unroll
    for (int t = 0; t < 2; ++t) {
        int d = threadIdx.x + t * 256;
        float m = mu[d]; v[t] = m;
        lsq += m * m;
        float s = sg[d];
        llog += __logf(s + EPS);
        lsig += s;
    }
    float sumsq  = block_sum<256>(lsq,  red);
    float sumlog = block_sum<256>(llog, red);
    float sumsig = block_sum<256>(lsig, red);

    float inv = 1.0f / fmaxf(sqrtf(sumsq), 1e-12f);
    float*    nf = (s2 ? n2f : n1f) + row * D;
    _Float16* nh = (s2 ? n2h : n1h) + row * D;
#pragma unroll
    for (int t = 0; t < 2; ++t) {
        int d = threadIdx.x + t * 256;
        float nv = v[t] * inv;
        nf[d] = nv;
        nh[d] = (_Float16)nv;
    }
    if (threadIdx.x == 0) {
        (s2 ? logdet2 : logdet1)[row] = sumlog;
        (s2 ? smean2  : smean1 )[row] = sumsig * (1.0f / (float)D);
        (s2 ? sq2     : sq1    )[row] = sumsq * inv * inv;
    }
}

// ------------------------------------- kernel 1: Bhattacharyya logits [B,B]
// One block per 16x16 (i,j) tile; LDS-stage 16 rows of sigma1/sigma2/n1/n2 in
// 128-wide d-chunks (pitch 132 -> conflict-free strided reads). TRANS-bound:
// one v_rcp_f32 + one v_log_f32 per (i,j,d).
#define DC    128
#define PITCH 132
__global__ void bd_logits_kernel(const float* __restrict__ sg1, const float* __restrict__ sg2,
                                 const float* __restrict__ n1,  const float* __restrict__ n2,
                                 const float* __restrict__ logdet1,
                                 const float* __restrict__ logdet2,
                                 const float* __restrict__ logit_scale,
                                 float* __restrict__ logits) {
    __shared__ float ls1[16 * PITCH], ls2[16 * PITCH], ln1[16 * PITCH], ln2[16 * PITCH];
    const int ti = threadIdx.x & 15;
    const int tj = threadIdx.x >> 4;
    const int i0 = blockIdx.x * 16;
    const int j0 = blockIdx.y * 16;

    float term1 = 0.f, ldet = 0.f;
    for (int d0 = 0; d0 < D; d0 += DC) {
        __syncthreads();
        for (int t = threadIdx.x; t < 16 * (DC / 4); t += 256) {
            int r = t >> 5;              // row in tile
            int c = (t & 31) << 2;       // float offset within chunk
            *(float4*)&ls1[r * PITCH + c] = *(const float4*)(sg1 + (i0 + r) * D + d0 + c);
            *(float4*)&ls2[r * PITCH + c] = *(const float4*)(sg2 + (j0 + r) * D + d0 + c);
            *(float4*)&ln1[r * PITCH + c] = *(const float4*)(n1  + (i0 + r) * D + d0 + c);
            *(float4*)&ln2[r * PITCH + c] = *(const float4*)(n2  + (j0 + r) * D + d0 + c);
        }
        __syncthreads();
        if (d0 + DC < D) {   // warm L2/WGP$ for the next chunk
            __builtin_prefetch(sg1 + (i0 + ti) * D + d0 + DC, 0, 0);
            __builtin_prefetch(n2  + (j0 + tj) * D + d0 + DC, 0, 0);
        }
#pragma unroll 8
        for (int d = 0; d < DC; ++d) {
            float sa   = 0.5f * (ls1[ti * PITCH + d] + ls2[tj * PITCH + d]) + EPS;
            float diff = ln1[ti * PITCH + d] - ln2[tj * PITCH + d];
            term1 += diff * diff * __builtin_amdgcn_rcpf(sa);
            ldet  += __logf(sa);
        }
    }
    float term2 = ldet - 0.5f * (logdet1[i0 + ti] + logdet2[j0 + tj]);
    float bd    = 0.125f * term1 + 0.5f * term2;
    float sim   = __expf(-bd * (1.0f / (float)D));
    logits[(i0 + ti) * B + (j0 + tj)] = logit_scale[0] * sim;
}

// ------------------------------------------- kernel 2: Gram = n1 @ n2^T (WMMA)
// One wave per 16x16 C-tile; 16x v_wmma_f32_16x16x32_f16 along K=512.
__device__ __forceinline__ v16h load_afrag(const _Float16* rowp, int lane) {
    // 16-bit A 16x32 layout: lanes<16 hold K {0..7, 16..23}; lanes>=16 {8..15, 24..31}
    const int koff = (lane & 16) ? 8 : 0;
    v8h lo = *(const v8h*)(rowp + koff);
    v8h hi = *(const v8h*)(rowp + koff + 16);
    return __builtin_shufflevector(lo, hi, 0,1,2,3,4,5,6,7,8,9,10,11,12,13,14,15);
}
__device__ __forceinline__ v16h load_bfrag(const _Float16* colp, int lane) {
    // B 32x16: lane holds column N=lane%16; lanes<16 K {0..7, 16..23}; lanes>=16 {8..15, 24..31}
    const int koff = (lane & 16) ? 8 : 0;
    v8h lo = *(const v8h*)(colp + koff);
    v8h hi = *(const v8h*)(colp + koff + 16);
    return __builtin_shufflevector(lo, hi, 0,1,2,3,4,5,6,7,8,9,10,11,12,13,14,15);
}

__global__ void gram_wmma_kernel(const _Float16* __restrict__ n1h,
                                 const _Float16* __restrict__ n2h,
                                 float* __restrict__ G) {
    const int lane = threadIdx.x;                 // 32 threads = 1 wave
    const int rowA = blockIdx.x * 16 + (lane & 15);
    const int rowB = blockIdx.y * 16 + (lane & 15);
    v8f c = {};
#pragma unroll
    for (int k = 0; k < D; k += 32) {
        v16h a = load_afrag(n1h + rowA * D + k, lane);
        v16h b = load_bfrag(n2h + rowB * D + k, lane);
        c = __builtin_amdgcn_wmma_f32_16x16x32_f16(false, a, false, b,
                                                   (short)0, c, false, false);
    }
    const int m0  = blockIdx.x * 16 + ((lane & 16) ? 8 : 0);
    const int col = blockIdx.y * 16 + (lane & 15);
#pragma unroll
    for (int r = 0; r < 8; ++r)
        G[(m0 + r) * B + col] = c[r];
}

// ---------------------- kernel 3: diagonal cross-entropy (rows and columns)
// blocks 0..511: rows of L; blocks 512..1023: rows of L^T (columns of L)
__global__ void ce_kernel(const float* __restrict__ logits,
                          float* __restrict__ ce_row, float* __restrict__ ce_col) {
    __shared__ float red[256];
    const int  b   = blockIdx.x;
    const bool col = b >= B;
    const int  r   = col ? b - B : b;
    const int  j0  = threadIdx.x, j1 = threadIdx.x + 256;
    float v0, v1;
    if (!col) { v0 = logits[r * B + j0]; v1 = logits[r * B + j1]; }
    else      { v0 = logits[j0 * B + r]; v1 = logits[j1 * B + r]; }
    float m   = block_max<256>(fmaxf(v0, v1), red);
    float sum = block_sum<256>(__expf(v0 - m) + __expf(v1 - m), red);
    if (threadIdx.x == 0) {
        float diag = logits[r * B + r];
        (col ? ce_col : ce_row)[r] = __logf(sum) + m - diag;
    }
}

// ----------------------- kernel 4: per-row uncertainty from the Gram matrix
__global__ void unc_kernel(const float* __restrict__ G,
                           const float* __restrict__ sq1, const float* __restrict__ sq2,
                           float* __restrict__ unc) {
    __shared__ float red[256];
    const int i = blockIdx.x;
    const float a = sq1[i];
    const int j0 = threadIdx.x, j1 = threadIdx.x + 256;
    float d0 = a + sq2[j0] - 2.0f * G[i * B + j0];
    float d1 = a + sq2[j1] - 2.0f * G[i * B + j1];
    if (j0 == i) d0 = INFINITY;   // mask the diagonal (off-diagonal max of sim)
    if (j1 == i) d1 = INFINITY;
    float m = block_min<256>(fminf(d0, d1), red);
    if (threadIdx.x == 0) {
        float dii = a + sq2[i] - 2.0f * G[i * B + i];
        // exp(-diag/neg_max) = exp(-exp(m - dii))
        unc[i] = __expf(-__expf(m - dii));
    }
}

// --------------------------------------- kernel 5: final scalar reductions
__global__ void final_kernel(const float* __restrict__ unc,
                             const float* __restrict__ smean1, const float* __restrict__ smean2,
                             const float* __restrict__ ce_row, const float* __restrict__ ce_col,
                             float* __restrict__ out) {
    __shared__ float red[512];
    const int t = threadIdx.x;
    float u = unc[t];
    float s = 0.5f * (smean1[t] + smean2[t]);
    float Su  = block_sum<512>(u,        red);
    float Suu = block_sum<512>(u * u,    red);
    float Sss = block_sum<512>(s * s,    red);
    float Sus = block_sum<512>(u * s,    red);
    float Sr  = block_sum<512>(ce_row[t], red);
    float Sc  = block_sum<512>(ce_col[t], red);
    if (t == 0) {
        float cs = Sus / (fmaxf(sqrtf(Suu), 1e-12f) * fmaxf(sqrtf(Sss), 1e-12f));
        out[0] = 0.5f * (Sr + Sc) * (1.0f / (float)B);   // loss_pro
        out[1] = 2.4f * (1.0f - cs);                     // loss_rank * 2.4
        out[2] = 0.5f * Su * (1.0f / (float)B);          // var_loss
    }
}

// ------------------------------------------------------------------ launcher
extern "C" void kernel_launch(void* const* d_in, const int* in_sizes, int n_in,
                              void* d_out, int out_size, void* d_ws, size_t ws_size,
                              hipStream_t stream) {
    const float* mu1 = (const float*)d_in[0];
    const float* sg1 = (const float*)d_in[1];
    const float* mu2 = (const float*)d_in[2];
    const float* sg2 = (const float*)d_in[3];
    const float* lsc = (const float*)d_in[4];
    float* out = (float*)d_out;

    // workspace carve-up (~5.3 MB)
    const size_t NBD = (size_t)B * D;
    float* n1f    = (float*)d_ws;
    float* n2f    = n1f + NBD;
    float* logits = n2f + NBD;
    float* G      = logits + NBD;
    _Float16* n1h = (_Float16*)(G + NBD);
    _Float16* n2h = n1h + NBD;
    float* small  = (float*)(n2h + NBD);
    float* logdet1 = small;
    float* logdet2 = small + 512;
    float* smean1  = small + 1024;
    float* smean2  = small + 1536;
    float* sq1     = small + 2048;
    float* sq2     = small + 2560;
    float* unc     = small + 3072;
    float* ce_row  = small + 3584;
    float* ce_col  = small + 4096;

    rowstats_kernel<<<dim3(2 * B), dim3(256), 0, stream>>>(
        mu1, sg1, mu2, sg2, n1f, n2f, n1h, n2h,
        logdet1, logdet2, smean1, smean2, sq1, sq2);

    bd_logits_kernel<<<dim3(B / 16, B / 16), dim3(256), 0, stream>>>(
        sg1, sg2, n1f, n2f, logdet1, logdet2, lsc, logits);

    gram_wmma_kernel<<<dim3(B / 16, B / 16), dim3(32), 0, stream>>>(n1h, n2h, G);

    ce_kernel<<<dim3(2 * B), dim3(256), 0, stream>>>(logits, ce_row, ce_col);

    unc_kernel<<<dim3(B), dim3(256), 0, stream>>>(G, sq1, sq2, unc);

    final_kernel<<<dim3(1), dim3(512), 0, stream>>>(
        unc, smean1, smean2, ce_row, ce_col, out);
}
